// QLSTM_65481071403391
// MI455X (gfx1250) — matrix operand order repaired and checked by reading, not compiled, not run on hardware
//
#include <hip/hip_runtime.h>
#include <hip/hip_bf16.h>

// QLSTM for MI455X (gfx1250, wave32).
// Per-wave: 16 batch rows across all T=512 steps, state in VGPRs.
// Gate GEMM = V_WMMA_F32_16X16X32_F16 (M=16 rows, K=32=D+H, N=16 gates).
// Attention GEMM = same WMMA with K zero-padded to 32.
// Transcendentals use v_cos_f32 / v_exp_f32 / v_rcp_f32 / v_tanh_f32 fast paths
// (no IEEE division sequences on the recurrence critical path).

typedef __attribute__((ext_vector_type(16))) _Float16 v16h;
typedef __attribute__((ext_vector_type(8)))  _Float16 v8h;
typedef __attribute__((ext_vector_type(8)))  float    v8f;

#define T_STEPS 512
#define BATCH   2048
#define DIMX    16
#define HID     16

__device__ __forceinline__ float fast_rcp(float v) {
    return __builtin_amdgcn_rcpf(v);           // v_rcp_f32
}
__device__ __forceinline__ float fast_sigmoid(float v) {
    // v = cos(...) in [-1,1]; 1+exp(-v) in [1.36,3.72] -> fast rcp is safe
    return fast_rcp(1.0f + __expf(-v));
}
__device__ __forceinline__ float fast_tanh(float v) {
#if __has_builtin(__builtin_amdgcn_tanhf)
    return __builtin_amdgcn_tanhf(v);          // v_tanh_f32 (CDNA5 TRANS op)
#else
    float e = __expf(-2.0f * v);
    return (1.0f - e) * fast_rcp(1.0f + e);
#endif
}

__global__ __launch_bounds__(32) void qlstm_wmma_kernel(
    const float* __restrict__ x,                                   // [T,B,D]
    const float* __restrict__ Wf, const float* __restrict__ bf, const float* __restrict__ thf,
    const float* __restrict__ Wi, const float* __restrict__ bi, const float* __restrict__ thi,
    const float* __restrict__ Wu, const float* __restrict__ bu, const float* __restrict__ thu,
    const float* __restrict__ Wo, const float* __restrict__ bo, const float* __restrict__ tho,
    const float* __restrict__ th_attn,
    const float* __restrict__ Wc, const float* __restrict__ bc,
    float* __restrict__ out)                                       // [T*B*H + B*H + B*H]
{
    // Per-wave hx relay tile (C/D layout -> A layout transform), 512 B.
    __shared__ _Float16 lds_h[16][16];     // [row m][col n], f16

    const int lane = threadIdx.x;          // 0..31 (wave32)
    const int g    = lane >> 4;            // lane half-group
    const int ml   = lane & 15;            // A: row m ; B/C: column n
    const int b0   = blockIdx.x * 16;      // batch tile origin

    // ---------------- Weights in B-layout (lane: N=ml, K=8g+j / 16+8g+j-8) ----------------
    // B[k][n] = W[n][k]  (lin = comb @ W.T)
    v16h wF, wI, wU, wO, wC;
    {
        const int kb = g * 8;
        const float* rf = Wf + ml * 32 + kb;
        const float* ri = Wi + ml * 32 + kb;
        const float* ru = Wu + ml * 32 + kb;
        const float* ro = Wo + ml * 32 + kb;
        #pragma unroll
        for (int j = 0; j < 16; ++j) {
            wF[j] = (_Float16)rf[j + (j >= 8 ? 8 : 0)];
            wI[j] = (_Float16)ri[j + (j >= 8 ? 8 : 0)];
            wU[j] = (_Float16)ru[j + (j >= 8 ? 8 : 0)];
            wO[j] = (_Float16)ro[j + (j >= 8 ? 8 : 0)];
            // attn B: Wc^T padded with zeros for K=16..31
            wC[j] = (j < 8) ? (_Float16)Wc[ml * 16 + 8 * g + j] : (_Float16)0.0f;
        }
    }

    // Per-lane gate bias (C column = gate index = ml)
    const float biasF = bf[ml] + thf[ml];
    const float biasI = bi[ml] + thi[ml];
    const float biasU = bu[ml] + thu[ml];
    const float biasO = bo[ml] + tho[ml];
    const float biasC = bc[ml];

    // th_attn slice matching this lane's x slice (features 8g .. 8g+7)
    float tha[8];
    #pragma unroll
    for (int j = 0; j < 8; ++j) tha[j] = th_attn[8 * g + j];

    // ---------------- Recurrent state in C/D layout: lane holds (m=r+8g, n=ml) ----
    float hx[8], cx[8];
    #pragma unroll
    for (int r = 0; r < 8; ++r) { hx[r] = 0.0f; cx[r] = 0.0f; }
    #pragma unroll
    for (int r = 0; r < 8; ++r) lds_h[r + 8 * g][ml] = (_Float16)0.0f;
    __syncthreads();   // single-wave WG: barrier lowers to S_NOP, waits only on DScnt

    // Lane's x pointer: row b0+ml, features 8g..8g+7 (two float4 = 32 B, coalesced)
    const float* xlane = x + ((size_t)b0 + (size_t)ml) * DIMX + 8 * g;
    const size_t xstep = (size_t)BATCH * DIMX;

    for (int t = 0; t < T_STEPS; ++t) {
        const float* xp = xlane + (size_t)t * xstep;
        const float4 x0 = *(const float4*)(xp);
        const float4 x1 = *(const float4*)(xp + 4);
        float xv[8] = { x0.x, x0.y, x0.z, x0.w, x1.x, x1.y, x1.z, x1.w };
        if (t + 1 < T_STEPS)
            __builtin_prefetch(xlane + (size_t)(t + 1) * xstep, 0, 0);

        // A (comb) per ISA 16-bit A 16x32 layout:
        //   a[0..7]  : K = 8g+j    -> x features
        //   a[8..15] : K = 16+8g+j -> hx features (from LDS relay)
        v16h a;
        #pragma unroll
        for (int j = 0; j < 8; ++j) a[j] = (_Float16)xv[j];
        {
            v8h hv = *(const v8h*)(&lds_h[ml][8 * g]);   // 16 B aligned ds_load
            #pragma unroll
            for (int j = 0; j < 8; ++j) a[8 + j] = hv[j];
        }
        // A (ctx) : ctx = cos(x + th_attn) in K=0..15, zeros in K=16..31
        v16h a2;
        #pragma unroll
        for (int j = 0; j < 8; ++j) a2[j] = (_Float16)__cosf(xv[j] + tha[j]);
        #pragma unroll
        for (int j = 0; j < 8; ++j) a2[8 + j] = (_Float16)0.0f;

        v8f z = {};
        v8f cF = __builtin_amdgcn_wmma_f32_16x16x32_f16(false, a,  false, wF, (short)0, z, false, false);
        v8f cI = __builtin_amdgcn_wmma_f32_16x16x32_f16(false, a,  false, wI, (short)0, z, false, false);
        v8f cU = __builtin_amdgcn_wmma_f32_16x16x32_f16(false, a,  false, wU, (short)0, z, false, false);
        v8f cO = __builtin_amdgcn_wmma_f32_16x16x32_f16(false, a,  false, wO, (short)0, z, false, false);
        v8f cA = __builtin_amdgcn_wmma_f32_16x16x32_f16(false, a2, false, wC, (short)0, z, false, false);

        __syncthreads();   // all lanes finished reading lds_h before we overwrite

        const size_t outOff = (size_t)t * ((size_t)BATCH * HID) + (size_t)b0 * HID + (size_t)ml;
        #pragma unroll
        for (int r = 0; r < 8; ++r) {
            const int m = r + 8 * g;
            float f = fast_sigmoid(__cosf(cF[r] + biasF));
            float i = fast_sigmoid(__cosf(cI[r] + biasI));
            float u = fast_tanh   (__cosf(cU[r] + biasU));
            float o = fast_sigmoid(__cosf(cO[r] + biasO));
            float c = f * cx[r] + i * u;
            float h = o * fast_tanh(c) + (cA[r] + biasC);
            cx[r] = c;
            hx[r] = h;
            out[outOff + (size_t)m * HID] = h;           // outputs[t, b0+m, ml]
            lds_h[m][ml] = (_Float16)h;                  // relay for next step's A
        }
        __syncthreads();
    }

    // Final hx, cx appended after outputs
    const size_t base = (size_t)T_STEPS * BATCH * HID;
    #pragma unroll
    for (int r = 0; r < 8; ++r) {
        const int m = r + 8 * g;
        out[base + (size_t)(b0 + m) * HID + ml]                        = hx[r];
        out[base + (size_t)BATCH * HID + (size_t)(b0 + m) * HID + ml]  = cx[r];
    }
}

extern "C" void kernel_launch(void* const* d_in, const int* in_sizes, int n_in,
                              void* d_out, int out_size, void* d_ws, size_t ws_size,
                              hipStream_t stream) {
    const float* x       = (const float*)d_in[0];
    const float* Wf      = (const float*)d_in[1];
    const float* bf      = (const float*)d_in[2];
    const float* thf     = (const float*)d_in[3];
    const float* Wi      = (const float*)d_in[4];
    const float* bi      = (const float*)d_in[5];
    const float* thi     = (const float*)d_in[6];
    const float* Wu      = (const float*)d_in[7];
    const float* bu      = (const float*)d_in[8];
    const float* thu     = (const float*)d_in[9];
    const float* Wo      = (const float*)d_in[10];
    const float* bo      = (const float*)d_in[11];
    const float* tho     = (const float*)d_in[12];
    const float* th_attn = (const float*)d_in[13];
    const float* Wc      = (const float*)d_in[14];
    const float* bc      = (const float*)d_in[15];
    float* out = (float*)d_out;

    dim3 grid(BATCH / 16);   // 128 waves, one 16-row batch tile each
    dim3 block(32);          // one wave32 per workgroup
    qlstm_wmma_kernel<<<grid, block, 0, stream>>>(
        x, Wf, bf, thf, Wi, bi, thi, Wu, bu, thu, Wo, bo, tho, th_attn, Wc, bc, out);
}
